// CSPNet_42279658062618
// MI455X (gfx1250) — compile-verified, matrix operand order checked by reading, not run on hardware
//
#include <hip/hip_runtime.h>

// ---------------------------------------------------------------------------
// CSPNet message-passing for MI455X (gfx1250), bf16 WMMA path.
//   N=50000 nodes, E=800000 edges, B=32 graphs, H=128, L=4 layers
//   edge MLP K: 325 padded to 352 (=11*32); node MLP K: 256 (=8*32)
//   Edge kernel: 32-edge M-tile per wave, async gather of h rows into LDS
//   (GLOBAL_LOAD_ASYNC_TO_LDS_B128, ASYNCcnt), 2 WMMAs per B-fragment load.
// ---------------------------------------------------------------------------

#define NN 50000
#define EE 800000
#define BBG 32
#define HH 128
#define LL 4
#define KE 352       // padded edge-MLP K
#define KE_REAL 325
#define KN 256
#define KB_E1 11
#define KB_E2 4
#define KB_N1 8
#define KB_N2 4
#define NB 8         // 128 / 16 output column blocks

typedef __attribute__((ext_vector_type(16))) __bf16 v16bf;
typedef __attribute__((ext_vector_type(8)))  float  v8f;
typedef __attribute__((ext_vector_type(4)))  unsigned int u32x4;
typedef __attribute__((ext_vector_type(8)))  unsigned int u32x8;

__device__ __forceinline__ unsigned short f2bf(float f) {
  unsigned int u = __builtin_bit_cast(unsigned int, f);
  u += 0x7FFFu + ((u >> 16) & 1u);           // round-to-nearest-even
  return (unsigned short)(u >> 16);
}
__device__ __forceinline__ float silu(float x) { return x / (1.0f + __expf(-x)); }

// Async memory->LDS copy, 16B per lane. VDST operand is a VGPR holding the
// LDS byte address (low 32 bits of the generic pointer, per ISA aperture rule).
// Tracked by ASYNCcnt; completed by s_wait_asynccnt 0.
__device__ __forceinline__ void async_ld_b128(unsigned lds_addr, const void* gaddr) {
  asm volatile("global_load_async_to_lds_b128 %0, %1, off"
               :: "v"(lds_addr), "v"(gaddr) : "memory");
}
__device__ __forceinline__ void wait_async0() {
  asm volatile("s_wait_asynccnt 0x0" ::: "memory");
}

// A-fragment per CDNA5 16-bit A layout (ISA 7.12.2):
//   lane l<16 : row l,   K = kbase+0..7  and kbase+16..23
//   lane l>=16: row l-16,K = kbase+8..15 and kbase+24..31
// caller passes row pointer (= &lds[row][0]); two 16B LDS loads per lane.
__device__ __forceinline__ v16bf a_frag(const unsigned short* row, int kbase, int lane) {
  const int koff = (lane >> 4) << 3;                  // 0 or 8
  u32x4 lo = *(const u32x4*)(row + kbase + koff);
  u32x4 hi = *(const u32x4*)(row + kbase + koff + 16);
  u32x8 f = {lo[0], lo[1], lo[2], lo[3], hi[0], hi[1], hi[2], hi[3]};
  return __builtin_bit_cast(v16bf, f);
}
// B-fragments are pre-packed per (kb, nb, lane) -> one contiguous 32B load.
__device__ __forceinline__ v16bf b_frag(const u32x8* wp, int kb, int nb, int lane) {
  return __builtin_bit_cast(v16bf, wp[(kb * NB + nb) * 32 + lane]);
}

// ---------------------------------------------------------------------------
// Prep kernels
// ---------------------------------------------------------------------------
__global__ void zero_k(float* __restrict__ p, int n) {
  for (int i = blockIdx.x * blockDim.x + threadIdx.x; i < n; i += gridDim.x * blockDim.x)
    p[i] = 0.0f;
}

__global__ void count_k(const int* __restrict__ ei, float* __restrict__ counts, int e) {
  int i = blockIdx.x * blockDim.x + threadIdx.x;
  if (i < e) atomicAdd(&counts[ei[i]], 1.0f);
}

__global__ void latip_k(const float* __restrict__ lat, float* __restrict__ latip) {
  int b = threadIdx.x;
  if (b >= BBG) return;
  #pragma unroll
  for (int i = 0; i < 3; ++i)
    #pragma unroll
    for (int k = 0; k < 3; ++k) {
      float s = 0.0f;
      #pragma unroll
      for (int j = 0; j < 3; ++j) s += lat[b * 9 + i * 3 + j] * lat[b * 9 + k * 3 + j];
      latip[b * 9 + i * 3 + k] = s;
    }
}

__global__ void init_h_k(const float* __restrict__ nf, float* __restrict__ h,
                         unsigned short* __restrict__ hbf, int n) {
  int i = blockIdx.x * blockDim.x + threadIdx.x;
  if (i < n) { float v = nf[i]; h[i] = v; hbf[i] = f2bf(v); }
}

// Convert f32 row-major W[L][K_real][128] into bf16 B-fragment layout:
//   dst flat index = (((l*KB + kb)*NB + nb)*32 + lane)*16 + i
//   element i<8  -> K = kb*32 + koff + i        (koff = (lane>>4)*8)
//   element i>=8 -> K = kb*32 + koff + 16 + (i-8)
//   column       = nb*16 + (lane&15); rows >= K_real are zero padding.
__global__ void pack_w_k(const float* __restrict__ W, unsigned short* __restrict__ dst,
                         int K_real, int KB, int total) {
  int idx = blockIdx.x * blockDim.x + threadIdx.x;
  if (idx >= total) return;
  int i = idx & 15;
  int t = idx >> 4;
  int lane = t & 31; t >>= 5;
  int nb = t & 7;    t >>= 3;
  int kb = t % KB;
  int l  = t / KB;
  int col  = nb * 16 + (lane & 15);
  int koff = (lane >> 4) * 8;
  int k = kb * 32 + koff + ((i < 8) ? i : (8 + i));
  unsigned short v = 0;
  if (k < K_real) v = f2bf(W[((size_t)l * K_real + k) * HH + col]);
  dst[idx] = v;
}

// ---------------------------------------------------------------------------
// Edge kernel: 1 wave = 32 edges (two 16-row WMMA subtiles, shared B frags).
// x = [hi|hj|latip(9)|sin(30)|cos(30)|pad]  (K=352)
// e = silu(silu(x@W1+b1)@W2+b2); atomic scatter-add into agg[ei].
// ---------------------------------------------------------------------------
__global__ __launch_bounds__(64) void edge_kernel(
    const unsigned short* __restrict__ hbf, const float* __restrict__ frac,
    const float* __restrict__ latip, const int* __restrict__ ei,
    const int* __restrict__ ej, const int* __restrict__ e2g,
    const u32x8* __restrict__ w1, const float* __restrict__ b1,
    const u32x8* __restrict__ w2, const float* __restrict__ b2,
    float* __restrict__ agg) {
  __shared__ __align__(16) unsigned short s_x[2][32][KE];   // 45056 B
  __shared__ __align__(16) unsigned short s_a[2][32][HH];   // 16384 B
  const int wv = threadIdx.x >> 5;
  const int lane = threadIdx.x & 31;
  const int tile = blockIdx.x * 2 + wv;                     // E/32 = 25000 exact

  // ---- async gather of hi|hj rows straight into LDS (no VGPR round-trip) ---
  // lanes 0..15 stage the hi half of a row, lanes 16..31 the hj half;
  // each lane moves 16B (8 bf16), one GLOBAL_LOAD_ASYNC_TO_LDS_B128 per row.
  {
    const int half = lane >> 4;                 // 0 = hi (ei), 1 = hj (ej)
    const int c8 = (lane & 15) * 8;             // bf16 column offset
    const int* idxp = half ? ej : ei;
    for (int m = 0; m < 32; ++m) {
      int node = idxp[tile * 32 + m];
      unsigned loff = (unsigned)(size_t)&s_x[wv][m][half * HH + c8];
      async_ld_b128(loff, &hbf[node * HH + c8]);
    }
  }
  // geometry features overlap with the in-flight async copies
  // (disjoint LDS bytes: columns 256..351): one edge per lane
  {
    int e = tile * 32 + lane;
    int ni = ei[e], nj = ej[e], g = e2g[e];
    unsigned short* xr = &s_x[wv][lane][0];
    float fd[3];
    #pragma unroll
    for (int d = 0; d < 3; ++d) {
      float df = frac[nj * 3 + d] - frac[ni * 3 + d];
      fd[d] = df - floorf(df);                              // mod 1.0
    }
    #pragma unroll
    for (int t = 0; t < 9; ++t) xr[2 * HH + t] = f2bf(latip[g * 9 + t]);
    #pragma unroll
    for (int d = 0; d < 3; ++d)
      #pragma unroll
      for (int f = 0; f < 10; ++f) {
        float ang = fd[d] * (6.283185307179586f * (float)f);
        xr[265 + d * 10 + f] = f2bf(__sinf(ang));
        xr[295 + d * 10 + f] = f2bf(__cosf(ang));
      }
    #pragma unroll
    for (int t = KE_REAL; t < KE; ++t) xr[t] = 0;           // K padding
  }
  wait_async0();                                            // asyncs landed
  __syncthreads();

  const int n0 = lane & 15, grp = lane >> 4;
  const unsigned short* arow0 = &s_x[wv][n0][0];            // subtile 0: rows 0..15
  const unsigned short* arow1 = &s_x[wv][16 + n0][0];       // subtile 1: rows 16..31

  // GEMM1: [32,352] x [352,128] -- one B load feeds two WMMAs
  v8f acc0[NB], acc1[NB];
  #pragma unroll
  for (int nb = 0; nb < NB; ++nb) {
    acc0[nb] = (v8f){0, 0, 0, 0, 0, 0, 0, 0};
    acc1[nb] = (v8f){0, 0, 0, 0, 0, 0, 0, 0};
  }
  for (int kb = 0; kb < KB_E1; ++kb) {
    if (kb + 1 < KB_E1)                                     // warm next K-block
      __builtin_prefetch(&w1[((kb + 1) * NB) * 32 + lane], 0, 3);
    v16bf a0 = a_frag(arow0, kb * 32, lane);
    v16bf a1 = a_frag(arow1, kb * 32, lane);
    #pragma unroll
    for (int nb = 0; nb < NB; ++nb) {
      v16bf b = b_frag(w1, kb, nb, lane);
      acc0[nb] = __builtin_amdgcn_wmma_f32_16x16x32_bf16(false, a0, false, b,
                                                         (short)0, acc0[nb], false, false);
      acc1[nb] = __builtin_amdgcn_wmma_f32_16x16x32_bf16(false, a1, false, b,
                                                         (short)0, acc1[nb], false, false);
    }
  }
  // bias + SiLU, write bf16 activation tile (D layout: M = r + grp*8, N = n0)
  #pragma unroll
  for (int nb = 0; nb < NB; ++nb) {
    float bias = b1[nb * 16 + n0];
    #pragma unroll
    for (int r = 0; r < 8; ++r) {
      int m = r + grp * 8;
      s_a[wv][m][nb * 16 + n0]      = f2bf(silu(acc0[nb][r] + bias));
      s_a[wv][16 + m][nb * 16 + n0] = f2bf(silu(acc1[nb][r] + bias));
    }
  }
  __syncthreads();

  // GEMM2: [32,128] x [128,128]
  const unsigned short* brow0 = &s_a[wv][n0][0];
  const unsigned short* brow1 = &s_a[wv][16 + n0][0];
  #pragma unroll
  for (int nb = 0; nb < NB; ++nb) {
    acc0[nb] = (v8f){0, 0, 0, 0, 0, 0, 0, 0};
    acc1[nb] = (v8f){0, 0, 0, 0, 0, 0, 0, 0};
  }
  for (int kb = 0; kb < KB_E2; ++kb) {
    if (kb + 1 < KB_E2)
      __builtin_prefetch(&w2[((kb + 1) * NB) * 32 + lane], 0, 3);
    v16bf a0 = a_frag(brow0, kb * 32, lane);
    v16bf a1 = a_frag(brow1, kb * 32, lane);
    #pragma unroll
    for (int nb = 0; nb < NB; ++nb) {
      v16bf b = b_frag(w2, kb, nb, lane);
      acc0[nb] = __builtin_amdgcn_wmma_f32_16x16x32_bf16(false, a0, false, b,
                                                         (short)0, acc0[nb], false, false);
      acc1[nb] = __builtin_amdgcn_wmma_f32_16x16x32_bf16(false, a1, false, b,
                                                         (short)0, acc1[nb], false, false);
    }
  }
  // bias + SiLU + scatter-mean numerator (atomic f32 adds)
  #pragma unroll
  for (int nb = 0; nb < NB; ++nb) {
    float bias = b2[nb * 16 + n0];
    #pragma unroll
    for (int r = 0; r < 8; ++r) {
      int m = r + grp * 8;
      int node0 = ei[tile * 32 + m];
      int node1 = ei[tile * 32 + 16 + m];
      atomicAdd(&agg[node0 * HH + nb * 16 + n0], silu(acc0[nb][r] + bias));
      atomicAdd(&agg[node1 * HH + nb * 16 + n0], silu(acc1[nb][r] + bias));
    }
  }
}

// ---------------------------------------------------------------------------
// Node kernel: 1 wave = 16 nodes. y=[h | agg/denom] (K=256)
// o = silu(silu(y@nw1+b1)@nw2+b2); h += o (f32 + bf16 shadow update)
// ---------------------------------------------------------------------------
__global__ __launch_bounds__(160) void node_kernel(
    float* __restrict__ h, unsigned short* __restrict__ hbf,
    const float* __restrict__ agg, const float* __restrict__ counts,
    const u32x8* __restrict__ w1, const float* __restrict__ b1,
    const u32x8* __restrict__ w2, const float* __restrict__ b2) {
  __shared__ __align__(16) unsigned short s_y[5][16][KN];   // 40960 B
  __shared__ __align__(16) unsigned short s_a[5][16][HH];   // 20480 B
  const int wv = threadIdx.x >> 5;
  const int lane = threadIdx.x & 31;
  const int tile = blockIdx.x * 5 + wv;                     // N/16 = 3125 exact

  for (int m = 0; m < 16; ++m) {
    int node = tile * 16 + m;
    float inv = 1.0f / fmaxf(counts[node], 1.0f);
    #pragma unroll
    for (int c = 0; c < 4; ++c) {
      int col = lane * 4 + c;
      s_y[wv][m][col]      = f2bf(h[node * HH + col]);
      s_y[wv][m][HH + col] = f2bf(agg[node * HH + col] * inv);
    }
  }
  __syncthreads();

  const int n0 = lane & 15, grp = lane >> 4;
  const unsigned short* arow = &s_y[wv][n0][0];

  v8f acc[NB];
  #pragma unroll
  for (int nb = 0; nb < NB; ++nb) acc[nb] = (v8f){0, 0, 0, 0, 0, 0, 0, 0};
  for (int kb = 0; kb < KB_N1; ++kb) {
    if (kb + 1 < KB_N1)
      __builtin_prefetch(&w1[((kb + 1) * NB) * 32 + lane], 0, 3);
    v16bf a = a_frag(arow, kb * 32, lane);
    #pragma unroll
    for (int nb = 0; nb < NB; ++nb) {
      v16bf b = b_frag(w1, kb, nb, lane);
      acc[nb] = __builtin_amdgcn_wmma_f32_16x16x32_bf16(false, a, false, b,
                                                        (short)0, acc[nb], false, false);
    }
  }
  #pragma unroll
  for (int nb = 0; nb < NB; ++nb) {
    float bias = b1[nb * 16 + n0];
    #pragma unroll
    for (int r = 0; r < 8; ++r) {
      int m = r + grp * 8;
      s_a[wv][m][nb * 16 + n0] = f2bf(silu(acc[nb][r] + bias));
    }
  }
  __syncthreads();

  const unsigned short* arow2 = &s_a[wv][n0][0];
  v8f acc2[NB];
  #pragma unroll
  for (int nb = 0; nb < NB; ++nb) acc2[nb] = (v8f){0, 0, 0, 0, 0, 0, 0, 0};
  for (int kb = 0; kb < KB_N2; ++kb) {
    if (kb + 1 < KB_N2)
      __builtin_prefetch(&w2[((kb + 1) * NB) * 32 + lane], 0, 3);
    v16bf a = a_frag(arow2, kb * 32, lane);
    #pragma unroll
    for (int nb = 0; nb < NB; ++nb) {
      v16bf b = b_frag(w2, kb, nb, lane);
      acc2[nb] = __builtin_amdgcn_wmma_f32_16x16x32_bf16(false, a, false, b,
                                                         (short)0, acc2[nb], false, false);
    }
  }
  #pragma unroll
  for (int nb = 0; nb < NB; ++nb) {
    float bias = b2[nb * 16 + n0];
    #pragma unroll
    for (int r = 0; r < 8; ++r) {
      int m = r + grp * 8;
      int node = tile * 16 + m;
      int col = nb * 16 + n0;
      float o = silu(acc2[nb][r] + bias);
      float hn = h[node * HH + col] + o;                    // residual
      h[node * HH + col] = hn;
      hbf[node * HH + col] = f2bf(hn);
    }
  }
}

// ---------------------------------------------------------------------------
// Host launcher
// ---------------------------------------------------------------------------
extern "C" void kernel_launch(void* const* d_in, const int* in_sizes, int n_in,
                              void* d_out, int out_size, void* d_ws, size_t ws_size,
                              hipStream_t stream) {
  (void)in_sizes; (void)n_in; (void)out_size; (void)ws_size;
  const float* nf   = (const float*)d_in[0];
  const float* frac = (const float*)d_in[1];
  const float* lat  = (const float*)d_in[2];
  const int*   eidx = (const int*)d_in[3];
  const int*   e2g  = (const int*)d_in[4];
  const float* ew1  = (const float*)d_in[5];
  const float* eb1  = (const float*)d_in[6];
  const float* ew2  = (const float*)d_in[7];
  const float* eb2  = (const float*)d_in[8];
  const float* nw1  = (const float*)d_in[9];
  const float* nb1  = (const float*)d_in[10];
  const float* nw2  = (const float*)d_in[11];
  const float* nb2  = (const float*)d_in[12];
  const int* ei = eidx;
  const int* ej = eidx + EE;
  float* h = (float*)d_out;

  // workspace layout (bytes)
  char* ws = (char*)d_ws;
  float*          agg   = (float*)(ws + 0);                 // 25,600,000
  float*          cnts  = (float*)(ws + 25600000);          //    200,000
  unsigned short* hbf   = (unsigned short*)(ws + 25800000); // 12,800,000
  float*          latip = (float*)(ws + 38600000);          //      1,152
  unsigned short* ew1p  = (unsigned short*)(ws + 38601152); //    360,448
  unsigned short* ew2p  = (unsigned short*)(ws + 38961600); //    131,072
  unsigned short* nw1p  = (unsigned short*)(ws + 39092672); //    262,144
  unsigned short* nw2p  = (unsigned short*)(ws + 39354816); //    131,072

  // prep (cheap, once per call)
  zero_k<<<1024, 256, 0, stream>>>(cnts, NN);
  count_k<<<(EE + 255) / 256, 256, 0, stream>>>(ei, cnts, EE);
  latip_k<<<1, 32, 0, stream>>>(lat, latip);
  init_h_k<<<(NN * HH + 255) / 256, 256, 0, stream>>>(nf, h, hbf, NN * HH);
  pack_w_k<<<(LL * KB_E1 * NB * 32 * 16 + 255) / 256, 256, 0, stream>>>(
      ew1, ew1p, KE_REAL, KB_E1, LL * KB_E1 * NB * 32 * 16);
  pack_w_k<<<(LL * KB_E2 * NB * 32 * 16 + 255) / 256, 256, 0, stream>>>(
      ew2, ew2p, HH, KB_E2, LL * KB_E2 * NB * 32 * 16);
  pack_w_k<<<(LL * KB_N1 * NB * 32 * 16 + 255) / 256, 256, 0, stream>>>(
      nw1, nw1p, KN, KB_N1, LL * KB_N1 * NB * 32 * 16);
  pack_w_k<<<(LL * KB_N2 * NB * 32 * 16 + 255) / 256, 256, 0, stream>>>(
      nw2, nw2p, HH, KB_N2, LL * KB_N2 * NB * 32 * 16);

  for (int l = 0; l < LL; ++l) {
    zero_k<<<2048, 256, 0, stream>>>(agg, NN * HH);
    edge_kernel<<<EE / 32 / 2, 64, 0, stream>>>(
        hbf, frac, latip, ei, ej, e2g,
        (const u32x8*)ew1p + (size_t)l * KB_E1 * NB * 32, eb1 + l * HH,
        (const u32x8*)ew2p + (size_t)l * KB_E2 * NB * 32, eb2 + l * HH, agg);
    node_kernel<<<NN / 16 / 5, 160, 0, stream>>>(
        h, hbf, agg, cnts,
        (const u32x8*)nw1p + (size_t)l * KB_N1 * NB * 32, nb1 + l * HH,
        (const u32x8*)nw2p + (size_t)l * KB_N2 * NB * 32, nb2 + l * HH);
  }
}
